// NetXtVLADConsensus_19507741459015
// MI455X (gfx1250) — matrix operand order repaired and checked by reading, not compiled, not run on hardware
//
#include <hip/hip_runtime.h>
#include <hip/hip_bf16.h>

// ---------------- problem constants ----------------
#define Bb   32
#define Ll   300
#define Dd   1024
#define Gg   8
#define Cc   64
#define DE   2048
#define GD   256
#define NTOK (Bb * Ll)          // 9600
#define KV   (Ll * Gg)          // 2400

typedef __attribute__((ext_vector_type(16))) __bf16 v16bf;
typedef __attribute__((ext_vector_type(8)))  float  v8f;
typedef __attribute__((ext_vector_type(2)))  float  v2f;
typedef __attribute__((ext_vector_type(4)))  int    v4i;

union V16U { v16bf v; uint4 q[2]; };
union BF4U { __bf16 h[4]; uint2 u; };

// ---- gfx1250 async copy-to-LDS (ASYNCcnt path), with safe fallback ----
#if defined(__has_builtin)
#  if __has_builtin(__builtin_amdgcn_global_load_async_to_lds_b128)
#    define USE_ASYNC_CP 1
#  else
#    define USE_ASYNC_CP 0
#  endif
#  if __has_builtin(__builtin_amdgcn_s_wait_asynccnt)
#    define HAVE_WAIT_ASYNC 1
#  else
#    define HAVE_WAIT_ASYNC 0
#  endif
#else
#  define USE_ASYNC_CP 0
#  define HAVE_WAIT_ASYNC 0
#endif

__device__ __forceinline__ void cp16_to_lds(__bf16* l, const __bf16* g) {
#if USE_ASYNC_CP
    __builtin_amdgcn_global_load_async_to_lds_b128(
        (__attribute__((address_space(1))) v4i*)g,
        (__attribute__((address_space(3))) v4i*)l, 0, 0);
#else
    *(uint4*)l = *(const uint4*)g;
#endif
}
__device__ __forceinline__ void cp_wait() {
#if USE_ASYNC_CP
#  if HAVE_WAIT_ASYNC
    __builtin_amdgcn_s_wait_asynccnt(0);
#  else
    asm volatile("s_wait_asynccnt 0" ::: "memory");
#  endif
#endif
}

__device__ __forceinline__ __bf16 f2bf(float f) {
    unsigned u = __builtin_bit_cast(unsigned, f);
    u += 0x7FFFu + ((u >> 16) & 1u);                 // round-to-nearest-even
    unsigned short h = (unsigned short)(u >> 16);
    return __builtin_bit_cast(__bf16, h);
}

__device__ __forceinline__ float wave_sum(float v) {
#pragma unroll
    for (int o = 16; o; o >>= 1) v += __shfl_xor(v, o, 32);
    return v;
}
__device__ __forceinline__ float wave_max(float v) {
#pragma unroll
    for (int o = 16; o; o >>= 1) v = fmaxf(v, __shfl_xor(v, o, 32));
    return v;
}
__device__ __forceinline__ float block_sum(float v, float* sm8) {
    v = wave_sum(v);
    int w = threadIdx.x >> 5, ln = threadIdx.x & 31;
    if (ln == 0) sm8[w] = v;
    __syncthreads();
    float r = 0.f;
#pragma unroll
    for (int i = 0; i < 8; ++i) r += sm8[i];
    __syncthreads();
    return r;
}

// ---------------- 1) per-token L2 norm + bf16 convert ----------------
__global__ void k_l2norm_bf16(const float* __restrict__ x, __bf16* __restrict__ xn) {
    __shared__ float sm8[8];
    size_t base = (size_t)blockIdx.x * Dd + threadIdx.x * 4;
    float4 v = *(const float4*)(x + base);
    float ss = block_sum(v.x * v.x + v.y * v.y + v.z * v.z + v.w * v.w, sm8);
    float inv = 1.f / fmaxf(sqrtf(ss), 1e-12f);
    BF4U o;
    o.h[0] = f2bf(v.x * inv); o.h[1] = f2bf(v.y * inv);
    o.h[2] = f2bf(v.z * inv); o.h[3] = f2bf(v.w * inv);
    *(uint2*)(xn + base) = o.u;
}

// ---------------- 2) f32 -> bf16 convert, 4-wide ----------------
__global__ void k_cvt_bf16_v4(const float* __restrict__ in, __bf16* __restrict__ out, int n4) {
    int i = blockIdx.x * 256 + threadIdx.x;
    if (i < n4) {
        float4 v = ((const float4*)in)[i];
        BF4U o;
        o.h[0] = f2bf(v.x); o.h[1] = f2bf(v.y); o.h[2] = f2bf(v.z); o.h[3] = f2bf(v.w);
        ((uint2*)out)[i] = o.u;
    }
}

// ---------------- 3) bf16 WMMA GEMM, LDS double-buffered, async staging ----------
// out[m,n] = sum_k A[m,k]*W[n,k] (+bias[n]).  Block tile 64x64, KT=64, 8 waves.
// Wave w: rows [(w&1)*32, +32) (2 accumulators), cols [(w>>1)*16, +16).
#define KT    64
#define LPAD  8
#define LSTR  (64 + LPAD)   // 72 bf16 = 144B row stride in LDS (16B aligned, bank-spread)

__global__ void k_gemm_bf16(const __bf16* __restrict__ A, const __bf16* __restrict__ W,
                            const float* __restrict__ bias,
                            float* __restrict__ Cout, __bf16* __restrict__ Cbf,
                            int Mrows, int Ncols, int K) {
    __shared__ __bf16 smA[2][64 * LSTR];
    __shared__ __bf16 smB[2][64 * LSTR];

    const int tid  = threadIdx.x;
    const int w    = tid >> 5, lane = tid & 31;
    const int hi   = lane >> 4, l15 = lane & 15;
    const int wm   = w & 1, wn = w >> 1;
    const int mblk = blockIdx.y * 64;
    const int nblk = blockIdx.x * 64;

    // cooperative copy mapping: 256 thr = 64 rows x 4 segs; each thread moves 2x16B per tile
    const int crow = tid >> 2;
    const int cseg = (tid & 3) * 8;
    int arow = mblk + crow; if (arow > Mrows - 1) arow = Mrows - 1;   // clamp (M=32 case)
    const __bf16* gA = A + (size_t)arow * K + cseg;
    const __bf16* gW = W + (size_t)(nblk + crow) * K + cseg;
    __bf16* lA0 = &smA[0][crow * LSTR + cseg];
    __bf16* lA1 = &smA[1][crow * LSTR + cseg];
    __bf16* lB0 = &smB[0][crow * LSTR + cseg];
    __bf16* lB1 = &smB[1][crow * LSTR + cseg];

    auto fill = [&](int buf, int k0) {
        __bf16* la = buf ? lA1 : lA0;
        __bf16* lb = buf ? lB1 : lB0;
        cp16_to_lds(la,      gA + k0);
        cp16_to_lds(la + 32, gA + k0 + 32);
        cp16_to_lds(lb,      gW + k0);
        cp16_to_lds(lb + 32, gW + k0 + 32);
    };

    v8f acc0 = {}, acc1 = {};
    fill(0, 0);
    int buf = 0;
    for (int k0 = 0; k0 < K; k0 += KT) {
        cp_wait();
        __syncthreads();                       // tile[buf] ready for all waves
        if (k0 + KT < K) fill(buf ^ 1, k0 + KT);

        const __bf16* bA0 = &smA[buf][(wm * 32 + l15) * LSTR];
        const __bf16* bA1 = &smA[buf][(wm * 32 + 16 + l15) * LSTR];
        const __bf16* bB  = &smB[buf][(wn * 16 + l15) * LSTR + hi * 8];
#pragma unroll
        for (int ks = 0; ks < KT; ks += 32) {
            V16U a0, a1, b;
            a0.q[0] = *(const uint4*)(bA0 + ks + hi * 8);
            a0.q[1] = *(const uint4*)(bA0 + ks + 16 + hi * 8);
            a1.q[0] = *(const uint4*)(bA1 + ks + hi * 8);
            a1.q[1] = *(const uint4*)(bA1 + ks + 16 + hi * 8);
            b.q[0]  = *(const uint4*)(bB + ks);
            b.q[1]  = *(const uint4*)(bB + ks + 16);
            acc0 = __builtin_amdgcn_wmma_f32_16x16x32_bf16(
                       false, a0.v, false, b.v, (short)0, acc0, false, false);
            acc1 = __builtin_amdgcn_wmma_f32_16x16x32_bf16(
                       false, a1.v, false, b.v, (short)0, acc1, false, false);
        }
        __syncthreads();                       // reads of tile[buf] done before its refill
        buf ^= 1;
    }

    int col = nblk + wn * 16 + l15;
    float bz = bias ? bias[col] : 0.f;
    int rb = mblk + wm * 32 + hi * 8;
#pragma unroll
    for (int r = 0; r < 8; ++r) {
        int row0 = rb + r, row1 = rb + 16 + r;
        if (row0 < Mrows) {
            float v = acc0[r] + bz;
            size_t idx = (size_t)row0 * Ncols + col;
            Cout[idx] = v;
            if (Cbf) Cbf[idx] = f2bf(v);
        }
        if (row1 < Mrows) {
            float v = acc1[r] + bz;
            size_t idx = (size_t)row1 * Ncols + col;
            Cout[idx] = v;
            if (Cbf) Cbf[idx] = f2bf(v);
        }
    }
}

// ---------------- 4) attention = sigmoid(xe . Wa^T + ba), wave per (n,g) ----------
__global__ void k_attention(const float* __restrict__ xe, const float* __restrict__ Wa,
                            const float* __restrict__ ba, float* __restrict__ att) {
    int n = blockIdx.x, g = threadIdx.x >> 5, lane = threadIdx.x & 31;
    const float* xr = xe + (size_t)n * DE;
    const float* wr = Wa + (size_t)g * DE;
    float s = 0.f;
    for (int i = lane; i < DE; i += 32) s += xr[i] * wr[i];
    s = wave_sum(s);
    if (lane == 0) att[n * Gg + g] = 1.f / (1.f + __expf(-(s + ba[g])));
}

// ---------------- 5) BN1 stats over sa.view(-1, 64): channel = k mod 64 ----------
__global__ void k_bn1_stats(const float* __restrict__ sa, float* __restrict__ mu,
                            float* __restrict__ rs) {
    __shared__ float sm8[8];
    int ch = blockIdx.x;
    float s = 0.f, sq = 0.f;
    for (int n = threadIdx.x; n < NTOK; n += 256) {
        const float* p = sa + (size_t)n * (Cc * Gg) + ch;
#pragma unroll
        for (int j = 0; j < 8; ++j) { float v = p[64 * j]; s += v; sq += v * v; }
    }
    s  = block_sum(s,  sm8);
    sq = block_sum(sq, sm8);
    if (threadIdx.x == 0) {
        const float invN = 1.f / (float)(NTOK * Gg);
        float m = s * invN;
        float var = sq * invN - m * m;
        mu[ch] = m;
        rs[ch] = rsqrtf(var + 1e-5f);
    }
}

// ---------------- 6) BN + softmax(axis=c) + attention gate, in place ----------
__global__ void k_softmax_act(float* __restrict__ sa, const float* __restrict__ mu,
                              const float* __restrict__ rs, const float* __restrict__ gam,
                              const float* __restrict__ bet, const float* __restrict__ att) {
    int n = blockIdx.x, g = threadIdx.x >> 5, lane = threadIdx.x & 31;
    size_t base = (size_t)n * (Cc * Gg);
    int k0 = (2 * lane) * Gg + g;
    int k1 = (2 * lane + 1) * Gg + g;
    int c0 = k0 & 63, c1 = k1 & 63;
    float v0 = (sa[base + k0] - mu[c0]) * rs[c0] * gam[c0] + bet[c0];
    float v1 = (sa[base + k1] - mu[c1]) * rs[c1] * gam[c1] + bet[c1];
    float m = wave_max(fmaxf(v0, v1));
    float e0 = __expf(v0 - m), e1 = __expf(v1 - m);
    float s = wave_sum(e0 + e1);
    float scale = att[n * Gg + g] / s;
    sa[base + k0] = e0 * scale;
    sa[base + k1] = e1 * scale;
}

// ---------------- 7) a_sum[b,c] = sum_{l,g} act ----------------
__global__ void k_asum(const float* __restrict__ act, float* __restrict__ asum) {
    __shared__ float sm8[8];
    int b = blockIdx.x >> 6, c = blockIdx.x & 63;
    float s = 0.f;
    for (int l = threadIdx.x; l < Ll; l += 256) {
        const float* p = act + ((size_t)(b * Ll + l)) * (Cc * Gg) + c * Gg;
#pragma unroll
        for (int g = 0; g < 8; ++g) s += p[g];
    }
    s = block_sum(s, sm8);
    if (threadIdx.x == 0) asum[blockIdx.x] = s;
}

// ---------------- 8) first_term via f32 WMMA 16x16x4 ----------------
// first_term[b,c,d] = sum_{k=(l,g)} act[b,l,c,g] * xe[b,l,g*256+d]
__global__ void k_first_term(const float* __restrict__ act, const float* __restrict__ xe,
                             float* __restrict__ ft) {
    int b = blockIdx.z;
    int w = threadIdx.x >> 5, lane = threadIdx.x & 31;
    int hi = lane >> 4, l15 = lane & 15;
    int ctile = blockIdx.y * 32 + (w & 1) * 16;
    int dtile = blockIdx.x * 64 + (w >> 1) * 16;
    int c = ctile + l15;
    int n = dtile + l15;
    const float* ab = act + (size_t)b * Ll * (Cc * Gg);
    const float* xb = xe  + (size_t)b * Ll * DE;

    v8f acc = {};
    for (int k0 = 0; k0 < KV; k0 += 4) {
        int ka = k0 + hi * 2;
        v2f a, bo;
        a.x = ab[((ka    ) >> 3) * (Cc * Gg) + c * Gg + ((ka    ) & 7)];
        a.y = ab[((ka + 1) >> 3) * (Cc * Gg) + c * Gg + ((ka + 1) & 7)];
        int kb = k0 + hi;
        bo.x = xb[((kb    ) >> 3) * DE + ((kb    ) & 7) * GD + n];
        bo.y = xb[((kb + 2) >> 3) * DE + ((kb + 2) & 7) * GD + n];
        acc = __builtin_amdgcn_wmma_f32_16x16x4_f32(
                  false, a, false, bo, (short)0, acc, false, false);
    }
    int row = ctile + hi * 8;
#pragma unroll
    for (int r = 0; r < 8; ++r)
        ft[(size_t)b * (Cc * GD) + (size_t)(row + r) * GD + n] = acc[r];
}

// ---------------- 9) vlad = l2norm(first - a_sum*centroid), in place ----------
__global__ void k_vlad_norm(float* __restrict__ ft, const float* __restrict__ asum,
                            const float* __restrict__ cent) {
    __shared__ float sm8[8];
    int b = blockIdx.x >> 6, c = blockIdx.x & 63, d = threadIdx.x;
    float* p = ft + (size_t)b * (Cc * GD) + (size_t)c * GD;
    float v = p[d] - asum[blockIdx.x] * cent[(size_t)c * GD + d];
    float ss = block_sum(v * v, sm8);
    p[d] = v / fmaxf(sqrtf(ss), 1e-12f);
}

// ---------------- 10) final BN (batch of 32) + bf16 convert ----------------
__global__ void k_fbn_cvt(const float* __restrict__ vlad, const float* __restrict__ fg,
                          const float* __restrict__ fb, __bf16* __restrict__ vb) {
    int j = blockIdx.x * 256 + threadIdx.x;
    float s = 0.f, sq = 0.f;
#pragma unroll 4
    for (int i = 0; i < Bb; ++i) {
        float v = vlad[(size_t)i * (Cc * GD) + j];
        s += v; sq += v * v;
    }
    float m = s * (1.f / Bb);
    float var = sq * (1.f / Bb) - m * m;
    float sc = fg[j] * rsqrtf(var + 1e-5f);
    float sh = fb[j] - m * sc;
#pragma unroll 4
    for (int i = 0; i < Bb; ++i)
        vb[(size_t)i * (Cc * GD) + j] = f2bf(vlad[(size_t)i * (Cc * GD) + j] * sc + sh);
}

// ================= host launcher =================
extern "C" void kernel_launch(void* const* d_in, const int* in_sizes, int n_in,
                              void* d_out, int out_size, void* d_ws, size_t ws_size,
                              hipStream_t stream) {
    const float* x    = (const float*)d_in[0];
    const float* We   = (const float*)d_in[1];
    const float* be   = (const float*)d_in[2];
    const float* Ws   = (const float*)d_in[3];
    const float* bn1g = (const float*)d_in[4];
    const float* bn1b = (const float*)d_in[5];
    const float* Wa   = (const float*)d_in[6];
    const float* ba   = (const float*)d_in[7];
    const float* cent = (const float*)d_in[8];
    const float* fg   = (const float*)d_in[9];
    const float* fb   = (const float*)d_in[10];
    const float* Wl   = (const float*)d_in[11];
    const float* bl   = (const float*)d_in[12];
    float* out = (float*)d_out;

    char* ws = (char*)d_ws;
    size_t off = 0;
    auto alloc = [&](size_t bytes) { char* p = ws + off; off += (bytes + 255) & ~(size_t)255; return p; };

    __bf16* xn_bf = (__bf16*)alloc((size_t)NTOK * Dd * 2);
    __bf16* We_bf = (__bf16*)alloc((size_t)DE * Dd * 2);
    __bf16* Ws_bf = (__bf16*)alloc((size_t)(Cc * Gg) * DE * 2);
    __bf16* Wl_bf = (__bf16*)alloc((size_t)1024 * (Cc * GD) * 2);
    float*  xe    = (float*) alloc((size_t)NTOK * DE * 4);
    __bf16* xe_bf = (__bf16*)alloc((size_t)NTOK * DE * 2);
    float*  sa    = (float*) alloc((size_t)NTOK * Cc * Gg * 4);   // becomes activation in place
    float*  att   = (float*) alloc((size_t)NTOK * Gg * 4);
    float*  bnmu  = (float*) alloc(64 * 4);
    float*  bnrs  = (float*) alloc(64 * 4);
    float*  asum  = (float*) alloc((size_t)Bb * Cc * 4);
    float*  ft    = (float*) alloc((size_t)Bb * Cc * GD * 4);     // first_term -> vlad in place
    __bf16* vb    = (__bf16*)alloc((size_t)Bb * Cc * GD * 2);

    // 1) L2 norm + bf16
    k_l2norm_bf16<<<NTOK, 256, 0, stream>>>(x, xn_bf);
    // 2) weight converts (all element counts divisible by 4)
    k_cvt_bf16_v4<<<(DE * Dd / 4 + 255) / 256, 256, 0, stream>>>(We, We_bf, DE * Dd / 4);
    k_cvt_bf16_v4<<<((Cc * Gg) * DE / 4 + 255) / 256, 256, 0, stream>>>(Ws, Ws_bf, (Cc * Gg) * DE / 4);
    k_cvt_bf16_v4<<<(1024 * (Cc * GD) / 4 + 255) / 256, 256, 0, stream>>>(Wl, Wl_bf, 1024 * (Cc * GD) / 4);
    // 3) xe = xn . We^T + be   (M=9600, N=2048, K=1024), fused bf16 copy
    k_gemm_bf16<<<dim3(DE / 64, NTOK / 64), 256, 0, stream>>>(xn_bf, We_bf, be, xe, xe_bf,
                                                              NTOK, DE, Dd);
    // 4) sa = xe . Ws^T        (M=9600, N=512, K=2048)
    k_gemm_bf16<<<dim3((Cc * Gg) / 64, NTOK / 64), 256, 0, stream>>>(xe_bf, Ws_bf, nullptr, sa, nullptr,
                                                                     NTOK, Cc * Gg, DE);
    // 5) attention
    k_attention<<<NTOK, 256, 0, stream>>>(xe, Wa, ba, att);
    // 6) BN1 stats
    k_bn1_stats<<<64, 256, 0, stream>>>(sa, bnmu, bnrs);
    // 7) BN + softmax + gate (in place -> activation)
    k_softmax_act<<<NTOK, 256, 0, stream>>>(sa, bnmu, bnrs, bn1g, bn1b, att);
    // 8) a_sum
    k_asum<<<Bb * Cc, 256, 0, stream>>>(sa, asum);
    // 9) first_term via f32 WMMA
    k_first_term<<<dim3(GD / 64, Cc / 32, Bb), 256, 0, stream>>>(sa, xe, ft);
    // 10) vlad residual + intra-norm
    k_vlad_norm<<<Bb * Cc, 256, 0, stream>>>(ft, asum, cent);
    // 11) final BN + bf16
    k_fbn_cvt<<<(Cc * GD) / 256, 256, 0, stream>>>(ft, fg, fb, vb);
    // 12) out = vlad_bn . Wl^T + bl  (M=32, N=1024, K=16384)
    k_gemm_bf16<<<dim3(1024 / 64, 1), 256, 0, stream>>>(vb, Wl_bf, bl, out, nullptr,
                                                        Bb, 1024, Cc * GD);

    (void)in_sizes; (void)n_in; (void)out_size; (void)ws_size;
}
